// VSE_35433480192666
// MI455X (gfx1250) — matrix operand order repaired and checked by reading, not compiled, not run on hardware
//
#include <hip/hip_runtime.h>
#include <hip/hip_bf16.h>

typedef __attribute__((ext_vector_type(16))) _Float16 v16h;
typedef __attribute__((ext_vector_type(8)))  float    v8f;

#define DEV static __device__ __forceinline__

// ---- WMMA fragment index helpers (CDNA5 16-bit 16x16x32 layouts) ----
// A (16xK): lane holds row m=lane&15; element e -> k offset:
DEV int ka_off(int e, int lane) {
  int r = e >> 1;
  (void)r;
  return (lane & 16) | e;
}
// B (Kx16): lane holds col n=lane&15; element e -> k = (lane&16)+e  (contiguous)

DEV v8f wmma16(v16h a, v16h b, v8f c) {
  return __builtin_amdgcn_wmma_f32_16x16x32_f16(false, a, false, b, (short)0, c, false, false);
}

DEV v8f zacc() {
  v8f c;
#pragma unroll
  for (int r = 0; r < 8; ++r) c[r] = 0.f;
  return c;
}
DEV v8f bacc(const float* bias, int n0, int lane) {
  float bv = bias[n0 + (lane & 15)];
  v8f c;
#pragma unroll
  for (int r = 0; r < 8; ++r) c[r] = bv;
  return c;
}

// ---- A loaders (M=16, K window [ks*32, ks*32+32)) ----
DEV v16h ldA32(const float* A, int ld, int ks, int lane) {
  const float* row = A + (lane & 15) * ld + ks * 32;
  v16h a;
#pragma unroll
  for (int e = 0; e < 16; ++e) a[e] = (_Float16)row[ka_off(e, lane)];
  return a;
}
DEV v16h ldA32_roll8(const float* A, int ld, int ks, int lane) {  // row (m+8)%16
  const float* row = A + (((lane & 15) + 8) & 15) * ld + ks * 32;
  v16h a;
#pragma unroll
  for (int e = 0; e < 16; ++e) a[e] = (_Float16)row[ka_off(e, lane)];
  return a;
}
DEV v16h ldA32_shift(const float* A, int ld, int ks, int lane, int shift) {  // zero-pad rows
  int m = (lane & 15) + shift;
  bool ok = (m >= 0) && (m < 16);
  const float* row = A + (ok ? m : 0) * ld + ks * 32;
  v16h a;
#pragma unroll
  for (int e = 0; e < 16; ++e) a[e] = ok ? (_Float16)row[ka_off(e, lane)] : (_Float16)0.f;
  return a;
}
DEV v16h ldA16(const _Float16* A, int ld, int ks, int lane) {
  const _Float16* row = A + (lane & 15) * ld + ks * 32;
  v16h a;
#pragma unroll
  for (int e = 0; e < 16; ++e) a[e] = row[ka_off(e, lane)];
  return a;
}
DEV v16h ldA32_k16(const float* A, int ld, int lane) {  // K=16 zero-padded to 32
  const float* row = A + (lane & 15) * ld;
  v16h a;
#pragma unroll
  for (int e = 0; e < 16; ++e) {
    int k = ka_off(e, lane);
    a[e] = (k < 16) ? (_Float16)row[k] : (_Float16)0.f;
  }
  return a;
}
DEV v16h ldA_cat(const float* A0, int ld0, const float* A1, int ld1, int ks, int lane) {
  int m = lane & 15;
  v16h a;
#pragma unroll
  for (int e = 0; e < 16; ++e) {
    int k = ks * 32 + ka_off(e, lane);
    a[e] = (_Float16)((k < 64) ? A0[m * ld0 + k] : A1[m * ld1 + (k - 64)]);
  }
  return a;
}

// ---- B loaders ----
// B[k][n] = W[n][k], W row-major [N][wld]  (computes x @ W^T)
DEV v16h ldBW(const float* W, int wld, int n0, int k0, int lane) {
  const float* row = W + (size_t)(n0 + (lane & 15)) * wld + k0 + (lane & 16);
  v16h b;
#pragma unroll
  for (int e = 0; e < 16; ++e) b[e] = (_Float16)row[e];
  return b;
}
// 3-tap conv weight [N][64][3]: B[k][n] = W[n][k][tap]
DEV v16h ldBW3(const float* W, int n0, int tap, int ks, int lane) {
  const float* row = W + (size_t)(n0 + (lane & 15)) * 192 + tap;
  int kb = ks * 32 + (lane & 16);
  v16h b;
#pragma unroll
  for (int e = 0; e < 16; ++e) b[e] = (_Float16)row[(kb + e) * 3];
  return b;
}
// B[k][n] = Kd[n][k]  (q @ k^T), Kd f16 row-major [16][ld]
DEV v16h ldBT16(const _Float16* Kd, int ld, int ks, int lane) {
  const _Float16* row = Kd + (lane & 15) * ld + ks * 32 + (lane & 16);
  v16h b;
#pragma unroll
  for (int e = 0; e < 16; ++e) b[e] = row[e];
  return b;
}
// B[k][n] = V[k][n0+n], V f16 row-major [16][ld], zero-pad k>=16
DEV v16h ldBV16(const _Float16* V, int ld, int n0, int lane) {
  int n = n0 + (lane & 15);
  v16h b;
#pragma unroll
  for (int e = 0; e < 16; ++e) {
    int k = (lane & 16) | e;
    b[e] = (k < 16) ? V[k * ld + n] : (_Float16)0.f;
  }
  return b;
}

// ---- D stores: element r -> (m = r + 8*(lane>>4), n = lane&15) ----
DEV void stD32(float* D, int ld, v8f d, int lane) {
  int n = lane & 15, mh = (lane >> 4) << 3;
#pragma unroll
  for (int r = 0; r < 8; ++r) D[(mh + r) * ld + n] = d[r];
}
DEV void stD16(_Float16* D, int ld, v8f d, int lane) {
  int n = lane & 15, mh = (lane >> 4) << 3;
#pragma unroll
  for (int r = 0; r < 8; ++r) D[(mh + r) * ld + n] = (_Float16)d[r];
}

DEV float gelu_exact(float x) { return 0.5f * x * (1.f + erff(x * 0.70710678118654752f)); }

DEV void ln16x64(const float* S, int lds, float* D, int ldd,
                 const float* g, const float* b, int tid) {
  if (tid < 16) {
    const float* row = S + tid * lds;
    float* drow = D + tid * ldd;
    float mean = 0.f;
    for (int k = 0; k < 64; ++k) mean += row[k];
    mean *= (1.f / 64.f);
    float var = 0.f;
    for (int k = 0; k < 64; ++k) { float d = row[k] - mean; var += d * d; }
    var *= (1.f / 64.f);
    float inv = rsqrtf(var + 1e-5f);
    for (int k = 0; k < 64; ++k) drow[k] = (row[k] - mean) * inv * g[k] + b[k];
  }
}

__global__ __launch_bounds__(128) void vse_fused(
    const float* __restrict__ x,
    const float* __restrict__ c11_w, const float* __restrict__ c11_b,
    const float* __restrict__ c12_w, const float* __restrict__ c12_b,
    const float* __restrict__ lse1_w, const float* __restrict__ lse1_b,
    const float* __restrict__ lse2_w, const float* __restrict__ lse2_b,
    const float* __restrict__ ln1_g, const float* __restrict__ ln1_b,
    const float* __restrict__ ln2_g, const float* __restrict__ ln2_b,
    const float* __restrict__ mlp1_w, const float* __restrict__ mlp1_b,
    const float* __restrict__ mlp2_w, const float* __restrict__ mlp2_b,
    const float* __restrict__ aln_g, const float* __restrict__ aln_b,
    const float* __restrict__ qkv_w, const float* __restrict__ qkv_b,
    const float* __restrict__ proj_w, const float* __restrict__ proj_b,
    const float* __restrict__ relpos,
    const float* __restrict__ fln_g, const float* __restrict__ fln_b,
    const float* __restrict__ ff1_w, const float* __restrict__ ff1_b,
    const float* __restrict__ ff2_w, const float* __restrict__ ff2_b,
    float* __restrict__ out) {
  __shared__ float XT[16][132];      // x^T [l][c]
  __shared__ float CX[16][68];       // conv branch [l][c]
  __shared__ float T1[16][68];
  __shared__ float H[16][68];        // trans h (pre-ln residual)
  __shared__ float Z[16][68];        // transformer state z
  __shared__ float S1[16][68];       // layernorm output scratch
  __shared__ _Float16 QH[4][16][72]; // per-wave q/k/v/out head tiles
  __shared__ _Float16 KH[4][16][72];
  __shared__ _Float16 VH[4][16][72];
  __shared__ _Float16 OH[4][16][72];
  __shared__ float DOTS[4][16][20];
  __shared__ float PP[4][16][68];    // per-wave proj partials
  __shared__ _Float16 FH[16][264];   // FFN hidden (post-activation)
  __shared__ float RES[16][132];

  const int tid = threadIdx.x;
  const int lane = tid & 31;
  const int wid = tid >> 5;
  const int b = blockIdx.x;
  const float* xb = x + (size_t)b * 2048;

  // ---- load x^T ----
  {
    int c = tid;
#pragma unroll
    for (int l = 0; l < 16; ++l) XT[l][c] = xb[c * 16 + l];
  }
  __syncthreads();

  // ---- c11 (1x1 conv == [16,128]x[128,128]) + ReLU, split into CX / H ----
#pragma unroll
  for (int t = 0; t < 2; ++t) {
    int nt = wid + t * 4;
    v8f acc = bacc(c11_b, nt * 16, lane);
#pragma unroll
    for (int ks = 0; ks < 4; ++ks)
      acc = wmma16(ldA32(&XT[0][0], 132, ks, lane), ldBW(c11_w, 128, nt * 16, ks * 32, lane), acc);
#pragma unroll
    for (int r = 0; r < 8; ++r) acc[r] = fmaxf(acc[r], 0.f);
    if (nt < 4) stD32(&CX[0][nt * 16], 68, acc, lane);
    else        stD32(&H[0][(nt - 4) * 16], 68, acc, lane);
  }
  __syncthreads();

  // ---- lse1: 3-tap conv 64->64 (+leaky 0.01) ----
  {
    int nt = wid;
    v8f acc = bacc(lse1_b, nt * 16, lane);
#pragma unroll
    for (int tap = 0; tap < 3; ++tap)
#pragma unroll
      for (int ks = 0; ks < 2; ++ks)
        acc = wmma16(ldA32_shift(&CX[0][0], 68, ks, lane, tap - 1),
                     ldBW3(lse1_w, nt * 16, tap, ks, lane), acc);
#pragma unroll
    for (int r = 0; r < 8; ++r) { float v = acc[r]; acc[r] = v > 0.f ? v : 0.01f * v; }
    stD32(&T1[0][nt * 16], 68, acc, lane);
  }
  __syncthreads();
  // ---- lse2 (+leaky) then conv_x = t + 2*conv_x ----
  {
    int nt = wid;
    v8f acc = bacc(lse2_b, nt * 16, lane);
#pragma unroll
    for (int tap = 0; tap < 3; ++tap)
#pragma unroll
      for (int ks = 0; ks < 2; ++ks)
        acc = wmma16(ldA32_shift(&T1[0][0], 68, ks, lane, tap - 1),
                     ldBW3(lse2_w, nt * 16, tap, ks, lane), acc);
    int n = lane & 15, mh = (lane >> 4) << 3;
#pragma unroll
    for (int r = 0; r < 8; ++r) {
      float v = acc[r]; v = v > 0.f ? v : 0.01f * v;
      int m = mh + r;
      CX[m][nt * 16 + n] = v + 2.f * CX[m][nt * 16 + n];
    }
  }
  __syncthreads();

  // ---- ln1: Z = LN(H) ----
  ln16x64(&H[0][0], 68, &Z[0][0], 68, ln1_g, ln1_b, tid);
  __syncthreads();

  // ================= transformer layers =================
  for (int li = 0; li < 2; ++li) {
    const float* qw = qkv_w + (size_t)li * 1536 * 64;
    const float* qb = qkv_b + (size_t)li * 1536;
    const float* pw = proj_w + (size_t)li * 64 * 512;
    const float* pb = proj_b + (size_t)li * 64;
    const float* rp = relpos + (size_t)li * 8 * 49;

    // pre-attn LN
    ln16x64(&Z[0][0], 68, &S1[0][0], 68, aln_g + li * 64, aln_b + li * 64, tid);
    __syncthreads();

    v8f pacc[4];
#pragma unroll
    for (int nt = 0; nt < 4; ++nt) pacc[nt] = (wid == 0) ? bacc(pb, nt * 16, lane) : zacc();

#pragma unroll
    for (int hh = 0; hh < 2; ++hh) {
      int head = wid * 2 + hh;
      // q/k/v for this head (input rolled by -8 tokens)
#pragma unroll
      for (int nt = 0; nt < 4; ++nt) {
        v8f aq = bacc(qb, head * 64 + nt * 16, lane);
        v8f ak = bacc(qb, 512 + head * 64 + nt * 16, lane);
        v8f av = bacc(qb, 1024 + head * 64 + nt * 16, lane);
#pragma unroll
        for (int ks = 0; ks < 2; ++ks) {
          v16h a = ldA32_roll8(&S1[0][0], 68, ks, lane);
          aq = wmma16(a, ldBW(qw,                      64, head * 64 + nt * 16, ks * 32, lane), aq);
          ak = wmma16(a, ldBW(qw + (size_t)512 * 64,   64, head * 64 + nt * 16, ks * 32, lane), ak);
          av = wmma16(a, ldBW(qw + (size_t)1024 * 64,  64, head * 64 + nt * 16, ks * 32, lane), av);
        }
        stD16(&QH[wid][0][nt * 16], 72, aq, lane);
        stD16(&KH[wid][0][nt * 16], 72, ak, lane);
        stD16(&VH[wid][0][nt * 16], 72, av, lane);
      }
      // dots = q k^T * scale + relpos, shifted-window mask
      {
        v8f dacc = zacc();
#pragma unroll
        for (int ks = 0; ks < 2; ++ks)
          dacc = wmma16(ldA16(&QH[wid][0][0], 72, ks, lane),
                        ldBT16(&KH[wid][0][0], 72, ks, lane), dacc);
        int n = lane & 15, mh = (lane >> 4) << 3;
#pragma unroll
        for (int r = 0; r < 8; ++r) {
          int m = mh + r;
          int ri = (m >> 2) - (n >> 2) + 3;
          int rj = (m & 3) - (n & 3) + 3;
          float bias = rp[(head * 7 + ri) * 7 + rj];
          bool msk = ((head >= 6) && (((m >> 2) < 2) != ((n >> 2) < 2))) ||
                     ((head & 1) && (((m & 3) < 2) != ((n & 3) < 2)));
          float v = dacc[r] * 0.125f + bias;
          DOTS[wid][m][n] = msk ? -1e9f : v;
        }
      }
      // softmax over rows (lanes 0..15 of this wave)
      if (lane < 16) {
        float* row = &DOTS[wid][lane][0];
        float mx = row[0];
        for (int j = 1; j < 16; ++j) mx = fmaxf(mx, row[j]);
        float s = 0.f;
        for (int j = 0; j < 16; ++j) { float e = __expf(row[j] - mx); row[j] = e; s += e; }
        float inv = 1.f / s;
        for (int j = 0; j < 16; ++j) row[j] *= inv;
      }
      // out_h = attn @ v  (K=16 zero-padded)
#pragma unroll
      for (int nt = 0; nt < 4; ++nt) {
        v8f acc = wmma16(ldA32_k16(&DOTS[wid][0][0], 20, lane),
                         ldBV16(&VH[wid][0][0], 72, nt * 16, lane), zacc());
        stD16(&OH[wid][0][nt * 16], 72, acc, lane);
      }
      // proj partial: out_h @ proj_w[:, head*64 : head*64+64]^T
#pragma unroll
      for (int nt = 0; nt < 4; ++nt)
#pragma unroll
        for (int ks = 0; ks < 2; ++ks)
          pacc[nt] = wmma16(ldA16(&OH[wid][0][0], 72, ks, lane),
                            ldBW(pw, 512, nt * 16, head * 64 + ks * 32, lane), pacc[nt]);
    }
#pragma unroll
    for (int nt = 0; nt < 4; ++nt) stD32(&PP[wid][0][nt * 16], 68, pacc[nt], lane);
    __syncthreads();
    // z += sum of wave partials (attention residual)
    for (int e = tid; e < 1024; e += 128) {
      int m = e >> 6, n = e & 63;
      Z[m][n] += PP[0][m][n] + PP[1][m][n] + PP[2][m][n] + PP[3][m][n];
    }
    __syncthreads();

    // ---- FFN ----
    ln16x64(&Z[0][0], 68, &S1[0][0], 68, fln_g + li * 64, fln_b + li * 64, tid);
    __syncthreads();
    const float* f1w = ff1_w + (size_t)li * 256 * 64;
    const float* f1b = ff1_b + (size_t)li * 256;
    const float* f2w = ff2_w + (size_t)li * 64 * 256;
    const float* f2b = ff2_b + (size_t)li * 64;
#pragma unroll
    for (int t = 0; t < 4; ++t) {
      int nt = wid * 4 + t;
      v8f acc = bacc(f1b, nt * 16, lane);
#pragma unroll
      for (int ks = 0; ks < 2; ++ks)
        acc = wmma16(ldA32(&S1[0][0], 68, ks, lane), ldBW(f1w, 64, nt * 16, ks * 32, lane), acc);
#pragma unroll
      for (int r = 0; r < 8; ++r) acc[r] = gelu_exact(acc[r]);
      stD16(&FH[0][nt * 16], 264, acc, lane);
    }
    __syncthreads();
    {
      int nt = wid;
      v8f acc = bacc(f2b, nt * 16, lane);
#pragma unroll
      for (int ks = 0; ks < 8; ++ks)
        acc = wmma16(ldA16(&FH[0][0], 264, ks, lane), ldBW(f2w, 256, nt * 16, ks * 32, lane), acc);
      int n = lane & 15, mh = (lane >> 4) << 3;
#pragma unroll
      for (int r = 0; r < 8; ++r) { int m = mh + r; Z[m][nt * 16 + n] += acc[r]; }
    }
    __syncthreads();
  }

  // ---- h = h + z; final MLP ----
  for (int e = tid; e < 1024; e += 128) { int m = e >> 6, n = e & 63; H[m][n] += Z[m][n]; }
  __syncthreads();
  ln16x64(&H[0][0], 68, &S1[0][0], 68, ln2_g, ln2_b, tid);
  __syncthreads();
#pragma unroll
  for (int t = 0; t < 4; ++t) {
    int nt = wid * 4 + t;
    v8f acc = bacc(mlp1_b, nt * 16, lane);
#pragma unroll
    for (int ks = 0; ks < 2; ++ks)
      acc = wmma16(ldA32(&S1[0][0], 68, ks, lane), ldBW(mlp1_w, 64, nt * 16, ks * 32, lane), acc);
#pragma unroll
    for (int r = 0; r < 8; ++r) acc[r] = gelu_exact(acc[r]);
    stD16(&FH[0][nt * 16], 264, acc, lane);
  }
  __syncthreads();
  {
    int nt = wid;
    v8f acc = bacc(mlp2_b, nt * 16, lane);
#pragma unroll
    for (int ks = 0; ks < 8; ++ks)
      acc = wmma16(ldA16(&FH[0][0], 264, ks, lane), ldBW(mlp2_w, 256, nt * 16, ks * 32, lane), acc);
    int n = lane & 15, mh = (lane >> 4) << 3;
#pragma unroll
    for (int r = 0; r < 8; ++r) { int m = mh + r; H[m][nt * 16 + n] += acc[r]; }
  }
  __syncthreads();

  // ---- c12 over concat(conv_x, trans_out) ----
#pragma unroll
  for (int t = 0; t < 2; ++t) {
    int nt = wid + t * 4;
    v8f acc = bacc(c12_b, nt * 16, lane);
#pragma unroll
    for (int ks = 0; ks < 4; ++ks)
      acc = wmma16(ldA_cat(&CX[0][0], 68, &H[0][0], 68, ks, lane),
                   ldBW(c12_w, 128, nt * 16, ks * 32, lane), acc);
    stD32(&RES[0][nt * 16], 132, acc, lane);
  }
  __syncthreads();

  // ---- out = x + res (back to [b][c][l]) ----
  {
    int c = tid;
    float* ob = out + (size_t)b * 2048;
#pragma unroll
    for (int l = 0; l < 16; ++l) ob[c * 16 + l] = xb[c * 16 + l] + RES[l][c];
  }
}

extern "C" void kernel_launch(void* const* d_in, const int* in_sizes, int n_in,
                              void* d_out, int out_size, void* d_ws, size_t ws_size,
                              hipStream_t stream) {
  (void)n_in; (void)out_size; (void)d_ws; (void)ws_size;
  const float* p[30];
  for (int i = 0; i < 30; ++i) p[i] = (const float*)d_in[i];
  int B = in_sizes[0] / (128 * 16);
  vse_fused<<<dim3(B), dim3(128), 0, stream>>>(
      p[0], p[1], p[2], p[3], p[4], p[5], p[6], p[7], p[8], p[9],
      p[10], p[11], p[12], p[13], p[14], p[15], p[16], p[17], p[18], p[19],
      p[20], p[21], p[22], p[23], p[24], p[25], p[26], p[27], p[28], p[29],
      (float*)d_out);
}
